// GlobalAttention_41618233099047
// MI455X (gfx1250) — matrix operand order repaired and checked by reading, not compiled, not run on hardware
//
#include <hip/hip_runtime.h>
#include <stdint.h>

// ---------------------------------------------------------------------------
// GlobalAttention pooling for MI455X (gfx1250, wave32, WMMA)
//   scores = tanh(x @ w1^T) @ w2^T          -> bf16 WMMA (f32 accum)
//   segment softmax over sorted batch ids    -> wave reductions
//   out[g] = attn_g^T @ x_g (f32, exact x)   -> VALU accumulation
// ---------------------------------------------------------------------------

typedef __attribute__((ext_vector_type(16))) __bf16          v16bf;
typedef __attribute__((ext_vector_type(8)))  float           v8f;
typedef __attribute__((ext_vector_type(8)))  unsigned short  v8us;

#define F_DIM 128
#define H_DIM 128
#define O_DIM 8

__device__ __forceinline__ unsigned short f2bf_bits(float f) {
  union { float f; uint32_t u; } v; v.f = f;
  uint32_t u = v.u;
  uint32_t r = u + 0x7FFFu + ((u >> 16) & 1u);   // round-to-nearest-even
  return (unsigned short)(r >> 16);
}

union Frag16 {
  v16bf v;
  unsigned short s[16];
  struct { v8us lo; v8us hi; } h;
};

// ---------------------------------------------------------------------------
// Kernel 1: per-node scores via two chained WMMA GEMMs.
// Block = 128 threads = 4 waves; each wave owns one 16-node tile per loop trip.
// LDS: w1 as bf16 (32KB) + w2 zero-padded to 16 rows (4KB) + per-wave h tile
// (4 x 4KB) = 52KB.
// B fragments are software-pipelined one step deep so ds_load_b128 overlaps
// the previous v_wmma instead of stalling on s_wait_dscnt 0.
// ---------------------------------------------------------------------------
__global__ __launch_bounds__(128) void ga_scores_kernel(
    const float* __restrict__ x, const float* __restrict__ w1,
    const float* __restrict__ w2, float* __restrict__ scores,
    int N, int numTiles)
{
  __shared__ alignas(32) unsigned short w1_lds[H_DIM * F_DIM];   // [h][f] bf16
  __shared__ alignas(32) unsigned short w2_lds[16 * H_DIM];      // [o pad16][h]
  __shared__ alignas(32) unsigned short h_lds[4][16 * H_DIM];    // per-wave [m][h]

  const int tid = threadIdx.x;

  // Stage weights once (f32 -> bf16).
  for (int i = tid; i < H_DIM * F_DIM; i += 128) w1_lds[i] = f2bf_bits(w1[i]);
  for (int i = tid; i < 16 * H_DIM; i += 128)
    w2_lds[i] = (i < O_DIM * H_DIM) ? f2bf_bits(w2[i]) : (unsigned short)0;
  __syncthreads();

  const int wave   = tid >> 5;
  const int lane   = tid & 31;
  const int r      = lane & 15;        // A row / B column within the 16-tile
  const int hiHalf = lane >> 4;        // 0: lanes 0-15, 1: lanes 16-31
  // CDNA5 16-bit A(16x32) layout: lane<16 holds K {0..7,16..23}, lane>=16 {8..15,24..31}
  const int kOffA0 = hiHalf ? 8 : 0;
  const int kOffA1 = kOffA0 + 16;
  // CDNA5 16-bit B(32x16) layout: same K order as A: lane<16 holds K {0..7,16..23}, lane>=16 {8..15,24..31}
  const int kOffB  = hiHalf ? 8 : 0;

  for (int tile = blockIdx.x * 4 + wave; tile < numTiles; tile += gridDim.x * 4) {
    const int m0 = tile * 16;
    int row = m0 + r; if (row >= N) row = N - 1;   // clamp (guarded at store)
    const float* xrow = x + (size_t)row * F_DIM;

    // ---- A fragments for the 4 K-steps (x row, f32 -> bf16) ----
    Frag16 afrag[4];
#pragma unroll
    for (int kb = 0; kb < 4; ++kb) {
      const int kb32 = kb * 32;
#pragma unroll
      for (int j = 0; j < 8; ++j) {
        afrag[kb].s[j]     = f2bf_bits(xrow[kb32 + kOffA0 + j]);
        afrag[kb].s[8 + j] = f2bf_bits(xrow[kb32 + kOffA1 + j]);
      }
    }

    // ---- GEMM1: h[16 x 128] = x_tile @ w1^T, 8 N-tiles x 4 K-steps ----
    // Flat step index i = nt*4 + kb; B fragment for step i+1 is loaded before
    // the WMMA of step i (one-deep pipeline).
    v8f acc[8];
    Frag16 bcur;
    {
      const int hidx0 = 0 * 16 + r;
      bcur.h.lo = *(const v8us*)&w1_lds[hidx0 * F_DIM + 0 * 32 + kOffB];
      bcur.h.hi = *(const v8us*)&w1_lds[hidx0 * F_DIM + 0 * 32 + kOffB + 16];
    }
#pragma unroll
    for (int nt = 0; nt < 8; ++nt) {
      v8f c = {0.f, 0.f, 0.f, 0.f, 0.f, 0.f, 0.f, 0.f};
#pragma unroll
      for (int kb = 0; kb < 4; ++kb) {
        // prefetch next B fragment (clamped so the last step reloads step 31)
        int nidx = nt * 4 + kb + 1; if (nidx > 31) nidx = 31;
        const int nnt = nidx >> 2, nkb = nidx & 3;
        const int nhidx = nnt * 16 + r;
        Frag16 bnext;
        bnext.h.lo = *(const v8us*)&w1_lds[nhidx * F_DIM + nkb * 32 + kOffB];
        bnext.h.hi = *(const v8us*)&w1_lds[nhidx * F_DIM + nkb * 32 + kOffB + 16];

        c = __builtin_amdgcn_wmma_f32_16x16x32_bf16(
                false, afrag[kb].v, false, bcur.v, (short)0, c, false, false);
        bcur = bnext;
      }
      acc[nt] = c;
    }

    // ---- tanh, then C-layout -> LDS (row-major bf16) for re-fragmenting ----
    // C layout: VGPR v, lanes 0-15 -> M=v ; lanes 16-31 -> M=v+8 ; N=r
#pragma unroll
    for (int nt = 0; nt < 8; ++nt) {
#pragma unroll
      for (int v = 0; v < 8; ++v) {
        const int m = v + 8 * hiHalf;
        h_lds[wave][m * H_DIM + nt * 16 + r] = f2bf_bits(tanhf(acc[nt][v]));
      }
    }
    // wave-private LDS region: no cross-wave barrier needed

    // ---- GEMM2: scores[16 x 8] = tanh(h) @ w2^T (w2 padded to 16 cols) ----
    v8f c2 = {0.f, 0.f, 0.f, 0.f, 0.f, 0.f, 0.f, 0.f};
#pragma unroll
    for (int kb = 0; kb < 4; ++kb) {
      const int kb32 = kb * 32;
      Frag16 a2;
      a2.h.lo = *(const v8us*)&h_lds[wave][r * H_DIM + kb32 + kOffA0];
      a2.h.hi = *(const v8us*)&h_lds[wave][r * H_DIM + kb32 + kOffA1];
      Frag16 b2;
      b2.h.lo = *(const v8us*)&w2_lds[r * H_DIM + kb32 + kOffB];
      b2.h.hi = *(const v8us*)&w2_lds[r * H_DIM + kb32 + kOffB + 16];
      c2 = __builtin_amdgcn_wmma_f32_16x16x32_bf16(
               false, a2.v, false, b2.v, (short)0, c2, false, false);
    }

    // ---- store scores[node][o]: lane column o = r, rows v + 8*hiHalf ----
    if (r < O_DIM) {
#pragma unroll
      for (int v = 0; v < 8; ++v) {
        const int node = m0 + v + 8 * hiHalf;
        if (node < N) scores[(size_t)node * O_DIM + r] = c2[v];
      }
    }
  }
}

// ---------------------------------------------------------------------------
// Kernel 2: per-graph segment softmax + attention-weighted pooling.
// One 256-thread block (8 waves) per graph. Wave w reduces head w (wave32
// shuffle reductions), then all threads accumulate out[g, o, f] with o split
// 2-way and f = tid & 127 (coalesced f32 x reads).
// ---------------------------------------------------------------------------
__device__ __forceinline__ int lower_bound_i32(const int* a, int n, int v) {
  int lo = 0, hi = n;
  while (lo < hi) { int mid = (lo + hi) >> 1; if (a[mid] < v) lo = mid + 1; else hi = mid; }
  return lo;
}

__global__ __launch_bounds__(256) void ga_pool_kernel(
    const float* __restrict__ x, const int* __restrict__ batch,
    const float* __restrict__ scores, float* __restrict__ out, int N)
{
  __shared__ int   seg[2];
  __shared__ float smax_sh[8];
  __shared__ float denom_sh[8];
  __shared__ float attn_sh[64][8];

  const int g   = blockIdx.x;
  const int tid = threadIdx.x;
  if (tid == 0)  seg[0] = lower_bound_i32(batch, N, g);
  if (tid == 32) seg[1] = lower_bound_i32(batch, N, g + 1);
  __syncthreads();
  const int s = seg[0], e = seg[1];

  const int head = tid >> 5;   // wave id == head id (8 waves, 8 heads)
  const int lane = tid & 31;

  // per-head segment max
  float m = -3.402823466e38f;
  for (int n = s + lane; n < e; n += 32)
    m = fmaxf(m, scores[(size_t)n * O_DIM + head]);
#pragma unroll
  for (int off = 16; off; off >>= 1) m = fmaxf(m, __shfl_xor(m, off, 32));
  if (lane == 0) smax_sh[head] = m;
  __syncthreads();

  // per-head exp-sum
  const float mh = smax_sh[head];
  float ss = 0.f;
  for (int n = s + lane; n < e; n += 32)
    ss += __expf(scores[(size_t)n * O_DIM + head] - mh);
#pragma unroll
  for (int off = 16; off; off >>= 1) ss += __shfl_xor(ss, off, 32);
  if (lane == 0) denom_sh[head] = ss;
  __syncthreads();

  // pooled accumulation: thread owns f = tid&127, heads og..og+3
  float a0 = 0.f, a1 = 0.f, a2 = 0.f, a3 = 0.f;
  const int f  = tid & 127;
  const int og = (tid >> 7) * 4;

  for (int base = s; base < e; base += 64) {
    const int cnt = (e - base < 64) ? (e - base) : 64;
    __syncthreads();                       // chunk buffer reuse guard
    for (int i = tid; i < cnt * 8; i += 256) {
      const int n = base + (i >> 3), o = i & 7;
      attn_sh[i >> 3][o] =
          __expf(scores[(size_t)n * O_DIM + o] - smax_sh[o]) / denom_sh[o];
    }
    __syncthreads();
    for (int j = 0; j < cnt; ++j) {
      const float xv = x[(size_t)(base + j) * F_DIM + f];   // 2nd use hits WGP$
      a0 += attn_sh[j][og + 0] * xv;
      a1 += attn_sh[j][og + 1] * xv;
      a2 += attn_sh[j][og + 2] * xv;
      a3 += attn_sh[j][og + 3] * xv;
    }
  }

  const size_t ob = (size_t)g * (O_DIM * F_DIM) + (size_t)og * F_DIM + f;
  out[ob]           = a0;   // empty graphs write zeros (d_out is poisoned)
  out[ob + F_DIM]   = a1;
  out[ob + 2*F_DIM] = a2;
  out[ob + 3*F_DIM] = a3;
}

// ---------------------------------------------------------------------------
extern "C" void kernel_launch(void* const* d_in, const int* in_sizes, int n_in,
                              void* d_out, int out_size, void* d_ws, size_t ws_size,
                              hipStream_t stream) {
  const float* x     = (const float*)d_in[0];
  const int*   batch = (const int*)d_in[1];
  const float* w1    = (const float*)d_in[2];
  const float* w2    = (const float*)d_in[3];
  // d_in[4] = num_graphs (device scalar); recover G from out_size instead.
  const int N = in_sizes[1];
  const int G = out_size / (O_DIM * F_DIM);

  float* scores = (float*)d_ws;                 // N * 8 floats scratch

  const int numTiles = (N + 15) / 16;           // 16-node WMMA tiles
  int grid1 = (numTiles + 3) / 4;               // 4 tiles (waves) per block
  if (grid1 > 1024) grid1 = 1024;               // persistent waves amortize w1 staging

  ga_scores_kernel<<<dim3(grid1), dim3(128), 0, stream>>>(
      x, w1, w2, scores, N, numTiles);
  ga_pool_kernel<<<dim3(G), dim3(256), 0, stream>>>(
      x, batch, scores, (float*)d_out, N);
}
